// RCNN_37245956390937
// MI455X (gfx1250) — matrix operand order repaired and checked by reading, not compiled, not run on hardware
//
#include <hip/hip_runtime.h>
#include <hip/hip_bf16.h>
#include <stdint.h>

// ---------------------------------------------------------------------------
// Faster R-CNN head for MI455X (gfx1250, wave32, WMMA 16x16x32 f16).
// - All GEMMs use v_wmma_f32_16x16x32_f16.
// - fc7 (300x4096x25088, 411MB fp32 weights) streams the weight matrix from
//   HBM exactly once: one block covers all 300 M-rows (5x64 sub-tiles),
//   split-K over grid.y for occupancy, fp32 partials reduced afterwards.
// - Tile staging uses GLOBAL_LOAD_ASYNC_TO_LDS_B128 when the builtin exists
//   (ASYNCcnt + s_wait_asynccnt), else falls back to VGPR bounce.
// ---------------------------------------------------------------------------

typedef __attribute__((ext_vector_type(16))) _Float16 v16h;
typedef __attribute__((ext_vector_type(8)))  _Float16 h8;
typedef __attribute__((ext_vector_type(8)))  float    v8f;
typedef __attribute__((ext_vector_type(4)))  int      v4i;

#define FH 50
#define FW 50
#define NPOS 2500
#define NANC 22500
#define TOPN_ 6000
#define KEEPN 300
#define NMSW 94          // ceil(6000/64)
#define SORTN 32768
#define CF 512
#define K1 768           // stem K  (3*16*16)
#define K2 4608          // rpn  K  (9*512)
#define POOLK 25088      // 512*49
#define FC7N 4096
#define FC7_KSPLIT 4
#define FC7_KCH (POOLK / FC7_KSPLIT)   // 6272, multiple of 32

#if defined(__AMDGCN__) && __has_builtin(__builtin_amdgcn_global_load_async_to_lds_b128) && __has_builtin(__builtin_amdgcn_s_wait_asynccnt)
#define HAVE_ASYNC_LDS 1
#else
#define HAVE_ASYNC_LDS 0
#endif

__device__ inline h8 h8zero() {
  h8 v;
#pragma unroll
  for (int e = 0; e < 8; ++e) v[e] = (_Float16)0.f;
  return v;
}

// 16-byte global -> LDS copy (async DMA on gfx1250 when available).
__device__ inline void copy16_g2l(void* dst_lds, const void* src_glob) {
#if HAVE_ASYNC_LDS
  __builtin_amdgcn_global_load_async_to_lds_b128(
      (__attribute__((address_space(1))) v4i*)src_glob,
      (__attribute__((address_space(3))) v4i*)dst_lds,
      0, 0);
#else
  *(h8*)dst_lds = *(const h8*)src_glob;
#endif
}

__device__ inline void g2l_wait() {
#if HAVE_ASYNC_LDS
  __builtin_amdgcn_s_wait_asynccnt(0);
#endif
}

// ---------------------------------------------------------------------------
// Generic fp16 WMMA GEMM: C[M][N] = A[M][K](f16) * B16[K][N](f16) (+bias,relu).
// Block 64x64, 256 threads = 8 waves; wave -> 16x32 C slab (two 16x16 frags).
// K%32==0, N%64==0; M guarded.
// ---------------------------------------------------------------------------
__global__ __launch_bounds__(256) void gemm_wmma_f16(
    const _Float16* __restrict__ A, int lda,
    const _Float16* __restrict__ B16,
    const float* __restrict__ bias, int relu,
    float* __restrict__ C32, _Float16* __restrict__ C16, int ldc,
    int M, int N, int K, int Nreal) {
  __shared__ __attribute__((aligned(16))) _Float16 sA[64][32 + 8];
  __shared__ __attribute__((aligned(16))) _Float16 sB[32][64 + 8];

  const int tid  = threadIdx.x;
  const int lane = tid & 31;
  const int wave = tid >> 5;
  const int wr   = wave & 3;
  const int wc   = wave >> 2;
  const int m0   = blockIdx.x * 64;
  const int n0   = blockIdx.y * 64;

  v8f acc0 = {};
  v8f acc1 = {};

  for (int k0 = 0; k0 < K; k0 += 32) {
    // ---- stage A tile (64x32 halves) ----
    {
      const int row = tid >> 2;
      const int cb  = (tid & 3) * 8;
      const int m   = m0 + row;
      if (m < M) copy16_g2l(&sA[row][cb], A + (size_t)m * lda + k0 + cb);
      else       *(h8*)&sA[row][cb] = h8zero();
    }
    // ---- stage B tile (32x64 halves) ----
    {
      const int row = tid >> 3;
      const int cb  = (tid & 7) * 8;
      copy16_g2l(&sB[row][cb], B16 + (size_t)(k0 + row) * N + n0 + cb);
    }
    g2l_wait();
    __syncthreads();

    // A fragment: lane -> M=lane&15; lane[4] selects K-half groups.
    const int arow = wr * 16 + (lane & 15);
    const int akb  = (lane & 16) ? 8 : 0;
    h8 alo = *(const h8*)&sA[arow][akb];
    h8 ahi = *(const h8*)&sA[arow][akb + 16];
    v16h af;
#pragma unroll
    for (int e = 0; e < 8; ++e) { af[e] = alo[e]; af[8 + e] = ahi[e]; }

    // B fragments: lane -> N=lane&15; lane[4] selects K-half groups.
    const int nb = wc * 32;
    const int bn = nb + (lane & 15);
    const int bkb = (lane & 16) ? 8 : 0;
    v16h bf0, bf1;
#pragma unroll
    for (int e = 0; e < 8; ++e) {
      bf0[e] = sB[bkb + e][bn]; bf0[8 + e] = sB[bkb + 16 + e][bn];
      bf1[e] = sB[bkb + e][bn + 16]; bf1[8 + e] = sB[bkb + 16 + e][bn + 16];
    }

    acc0 = __builtin_amdgcn_wmma_f32_16x16x32_f16(false, af, false, bf0,
                                                  (short)0, acc0, false, false);
    acc1 = __builtin_amdgcn_wmma_f32_16x16x32_f16(false, af, false, bf1,
                                                  (short)0, acc1, false, false);
    __syncthreads();
  }

  const int mbase = m0 + wr * 16 + ((lane & 16) ? 8 : 0);
  const int nA    = n0 + wc * 32 + (lane & 15);
  const int nB    = nA + 16;
#pragma unroll
  for (int v = 0; v < 8; ++v) {
    const int m = mbase + v;
    if (m < M) {
      if (nA < Nreal) {
        float y = acc0[v] + (bias ? bias[nA] : 0.f);
        if (relu) y = fmaxf(y, 0.f);
        if (C32) C32[(size_t)m * ldc + nA] = y;
        if (C16) C16[(size_t)m * ldc + nA] = (_Float16)y;
      }
      if (nB < Nreal) {
        float y = acc1[v] + (bias ? bias[nB] : 0.f);
        if (relu) y = fmaxf(y, 0.f);
        if (C32) C32[(size_t)m * ldc + nB] = y;
        if (C16) C16[(size_t)m * ldc + nB] = (_Float16)y;
      }
    }
  }
}

// ---------------------------------------------------------------------------
// fc7 GEMM: Cp[kc][320][4096] = partial(pool[300x25088] * w_fc7^T); weights
// streamed fp32->f16 exactly once. Block covers ALL 300 rows (5x64 sub-tiles)
// x 64 cols; grid = (4096/64 N-tiles, FC7_KSPLIT K-chunks).
// ---------------------------------------------------------------------------
__global__ __launch_bounds__(256) void gemm_fc7(
    const _Float16* __restrict__ A,   // pool [300][25088]
    const float* __restrict__ Bt,     // w_fc7 [4096][25088]
    float* __restrict__ Cp) {         // [FC7_KSPLIT][320][4096]
  __shared__ __attribute__((aligned(16))) _Float16 sA[320][32 + 8];
  __shared__ __attribute__((aligned(16))) _Float16 sB[32][64 + 8];

  const int tid  = threadIdx.x;
  const int lane = tid & 31;
  const int wave = tid >> 5;
  const int wr   = wave & 3;
  const int wc   = wave >> 2;
  const int n0   = blockIdx.x * 64;
  const int kc   = blockIdx.y;
  const int kbeg = kc * FC7_KCH;
  const int kend = kbeg + FC7_KCH;

  v8f zero = {};
  v8f acc[5][2];
#pragma unroll
  for (int s = 0; s < 5; ++s) { acc[s][0] = zero; acc[s][1] = zero; }

  for (int k0 = kbeg; k0 < kend; k0 += 32) {
    // ---- stage A super-tile (320x32 halves; rows >=300 zero) ----
    {
      const int cb = (tid & 3) * 8;
#pragma unroll
      for (int s = 0; s < 5; ++s) {
        const int row = (tid >> 2) + 64 * s;
        if (row < KEEPN) copy16_g2l(&sA[row][cb], A + (size_t)row * POOLK + k0 + cb);
        else             *(h8*)&sA[row][cb] = h8zero();
      }
    }
    // ---- stage B tile: stream fp32 weights, convert to f16 in-register ----
    {
      const int nrow = tid >> 2;
      const int kb   = (tid & 3) * 8;
      const float* src = Bt + (size_t)(n0 + nrow) * POOLK + k0 + kb;
      __builtin_prefetch(src + 256, 0, 1);  // global_prefetch_b8
      float4 v0 = *(const float4*)src;
      float4 v1 = *(const float4*)(src + 4);
      sB[kb + 0][nrow] = (_Float16)v0.x;
      sB[kb + 1][nrow] = (_Float16)v0.y;
      sB[kb + 2][nrow] = (_Float16)v0.z;
      sB[kb + 3][nrow] = (_Float16)v0.w;
      sB[kb + 4][nrow] = (_Float16)v1.x;
      sB[kb + 5][nrow] = (_Float16)v1.y;
      sB[kb + 6][nrow] = (_Float16)v1.z;
      sB[kb + 7][nrow] = (_Float16)v1.w;
    }
    g2l_wait();
    __syncthreads();

    // B fragments loaded once, reused across the 5 M sub-tiles.
    const int nb = wc * 32;
    const int bn = nb + (lane & 15);
    const int bkb = (lane & 16) ? 8 : 0;
    v16h bf0, bf1;
#pragma unroll
    for (int e = 0; e < 8; ++e) {
      bf0[e] = sB[bkb + e][bn]; bf0[8 + e] = sB[bkb + 16 + e][bn];
      bf1[e] = sB[bkb + e][bn + 16]; bf1[8 + e] = sB[bkb + 16 + e][bn + 16];
    }

    const int akb = (lane & 16) ? 8 : 0;
#pragma unroll
    for (int s = 0; s < 5; ++s) {
      const int arow = s * 64 + wr * 16 + (lane & 15);
      h8 alo = *(const h8*)&sA[arow][akb];
      h8 ahi = *(const h8*)&sA[arow][akb + 16];
      v16h af;
#pragma unroll
      for (int e = 0; e < 8; ++e) { af[e] = alo[e]; af[8 + e] = ahi[e]; }
      acc[s][0] = __builtin_amdgcn_wmma_f32_16x16x32_f16(false, af, false, bf0,
                                                         (short)0, acc[s][0], false, false);
      acc[s][1] = __builtin_amdgcn_wmma_f32_16x16x32_f16(false, af, false, bf1,
                                                         (short)0, acc[s][1], false, false);
    }
    __syncthreads();
  }

  const int nA = n0 + wc * 32 + (lane & 15);
  const int nB = nA + 16;
  float* slab = Cp + (size_t)kc * 320 * FC7N;
#pragma unroll
  for (int s = 0; s < 5; ++s) {
    const int mbase = s * 64 + wr * 16 + ((lane & 16) ? 8 : 0);
#pragma unroll
    for (int v = 0; v < 8; ++v) {
      const int m = mbase + v;
      if (m < KEEPN) {
        slab[(size_t)m * FC7N + nA] = acc[s][0][v];
        slab[(size_t)m * FC7N + nB] = acc[s][1][v];
      }
    }
  }
}

__global__ void k_fc7_reduce(const float* __restrict__ Cp, const float* __restrict__ bias,
                             _Float16* __restrict__ out) {
  int i = blockIdx.x * blockDim.x + threadIdx.x;
  if (i >= KEEPN * FC7N) return;
  int m = i / FC7N, n = i % FC7N;
  float s = bias[n];
#pragma unroll
  for (int kc = 0; kc < FC7_KSPLIT; ++kc)
    s += Cp[((size_t)kc * 320 + m) * FC7N + n];
  out[i] = (_Float16)fmaxf(s, 0.f);
}

// ---------------------------------------------------------------------------
// Prep / glue kernels
// ---------------------------------------------------------------------------
__global__ void k_im2col1(const float* __restrict__ img, _Float16* __restrict__ A1) {
  int i = blockIdx.x * blockDim.x + threadIdx.x;
  if (i >= NPOS * K1) return;
  int p = i / K1, k = i % K1;
  int c = k / 256, r = (k % 256) / 16, s = k % 16;
  int y = p / FW, x = p % FW;
  A1[i] = (_Float16)img[(size_t)c * 640000 + (size_t)(y * 16 + r) * 800 + (x * 16 + s)];
}

__global__ void k_wfeatT(const float* __restrict__ w, _Float16* __restrict__ o) {
  int i = blockIdx.x * blockDim.x + threadIdx.x;
  if (i >= K1 * CF) return;
  int k = i / CF, n = i % CF;
  o[i] = (_Float16)w[(size_t)n * K1 + k];
}

__global__ void k_wrpnT(const float* __restrict__ w, _Float16* __restrict__ o) {
  int i = blockIdx.x * blockDim.x + threadIdx.x;
  if (i >= K2 * CF) return;
  int k = i / CF, oc = i % CF;
  int t = k / CF;
  int ci = k % CF;
  o[i] = (_Float16)w[(size_t)oc * K2 + (size_t)ci * 9 + t];
}

__global__ void k_wsd(const float* __restrict__ ws, const float* __restrict__ wd,
                      const float* __restrict__ bs, const float* __restrict__ bd,
                      _Float16* __restrict__ o, float* __restrict__ bias) {
  int i = blockIdx.x * blockDim.x + threadIdx.x;
  if (i < 64) bias[i] = (i < 18) ? bs[i] : ((i < 54) ? bd[i - 18] : 0.f);
  if (i >= CF * 64) return;
  int k = i / 64, n = i % 64;
  float v = 0.f;
  if (n < 18) v = ws[(size_t)n * CF + k];
  else if (n < 54) v = wd[(size_t)(n - 18) * CF + k];
  o[i] = (_Float16)v;
}

__global__ void k_wcr(const float* __restrict__ wc, const float* __restrict__ wr,
                      const float* __restrict__ bc, const float* __restrict__ br,
                      _Float16* __restrict__ o, float* __restrict__ bias) {
  int i = blockIdx.x * blockDim.x + threadIdx.x;
  if (i < 128) bias[i] = (i < 21) ? bc[i] : ((i < 105) ? br[i - 21] : 0.f);
  if (i >= FC7N * 128) return;
  int k = i / 128, n = i % 128;
  float v = 0.f;
  if (n < 21) v = wc[(size_t)n * FC7N + k];
  else if (n < 105) v = wr[(size_t)(n - 21) * FC7N + k];
  o[i] = (_Float16)v;
}

__global__ void k_featT(const float* __restrict__ pc, float* __restrict__ chw) {
  int i = blockIdx.x * blockDim.x + threadIdx.x;
  if (i >= CF * NPOS) return;
  int c = i / NPOS, p = i % NPOS;
  chw[i] = pc[(size_t)p * CF + c];
}

__global__ void k_im2col2(const float* __restrict__ pc, _Float16* __restrict__ A2) {
  int i = blockIdx.x * blockDim.x + threadIdx.x;
  if (i >= NPOS * K2) return;
  int p = i / K2, k = i % K2;
  int t = k / CF, ci = k % CF;
  int y = p / FW + t / 3 - 1;
  int x = p % FW + t % 3 - 1;
  float v = 0.f;
  if (y >= 0 && y < FH && x >= 0 && x < FW) v = pc[(size_t)(y * FW + x) * CF + ci];
  A2[i] = (_Float16)v;
}

__global__ void k_rpn_post(const float* __restrict__ scdl, float* __restrict__ fg,
                           float* __restrict__ boxes) {
  int a = blockIdx.x * blockDim.x + threadIdx.x;
  if (a >= NANC) return;
  int p = a / 9, j = a % 9;
  const float* row = scdl + (size_t)p * 64;
  float l0 = row[2 * j], l1 = row[2 * j + 1];
  fg[a] = 1.f / (1.f + expf(l0 - l1));
  float d0 = row[18 + 4 * j + 0], d1 = row[18 + 4 * j + 1];
  float d2 = row[18 + 4 * j + 2], d3 = row[18 + 4 * j + 3];
  int x = p % FW, y = p / FW;
  float cx = (x + 0.5f) * 16.f, cy = (y + 0.5f) * 16.f;
  int r = j / 3, s = j % 3;
  float rs = (r == 0) ? 0.70710678f : ((r == 1) ? 1.f : 1.41421356f);
  float sc = (s == 0) ? 8.f : ((s == 1) ? 16.f : 32.f);
  float aw = 16.f * sc / rs, ah = 16.f * sc * rs;
  float bcx = cx + d0 * aw, bcy = cy + d1 * ah;
  float bw = aw * expf(d2), bh = ah * expf(d3);
  boxes[a * 4 + 0] = fminf(fmaxf(bcx - 0.5f * bw, 0.f), 800.f);
  boxes[a * 4 + 1] = fminf(fmaxf(bcy - 0.5f * bh, 0.f), 800.f);
  boxes[a * 4 + 2] = fminf(fmaxf(bcx + 0.5f * bw, 0.f), 800.f);
  boxes[a * 4 + 3] = fminf(fmaxf(bcy + 0.5f * bh, 0.f), 800.f);
}

__global__ void k_sort_init(const float* __restrict__ fg, float* __restrict__ key,
                            int* __restrict__ idx) {
  int i = blockIdx.x * blockDim.x + threadIdx.x;
  if (i >= SORTN) return;
  key[i] = (i < NANC) ? fg[i] : -__builtin_inff();
  idx[i] = i;
}

__global__ void k_bitonic(float* __restrict__ key, int* __restrict__ idx, int j, int k) {
  int i = blockIdx.x * blockDim.x + threadIdx.x;
  int ixj = i ^ j;
  if (ixj <= i || i >= SORTN) return;
  bool up = ((i & k) == 0);
  float ki = key[i], kj = key[ixj];
  bool sw = up ? (ki < kj) : (ki > kj);
  if (sw) {
    key[i] = kj; key[ixj] = ki;
    int t = idx[i]; idx[i] = idx[ixj]; idx[ixj] = t;
  }
}

__global__ void k_gather_cand(const float* __restrict__ boxes, const int* __restrict__ idx,
                              float* __restrict__ cand) {
  int i = blockIdx.x * blockDim.x + threadIdx.x;
  if (i >= TOPN_) return;
  int s = idx[i];
  cand[i * 4 + 0] = boxes[s * 4 + 0];
  cand[i * 4 + 1] = boxes[s * 4 + 1];
  cand[i * 4 + 2] = boxes[s * 4 + 2];
  cand[i * 4 + 3] = boxes[s * 4 + 3];
}

__global__ __launch_bounds__(64) void k_nms_mask(const float* __restrict__ cand,
                                                 unsigned long long* __restrict__ mask) {
  __shared__ float bx[64][4];
  int i = blockIdx.x * 64 + threadIdx.x;
  int j0 = blockIdx.y * 64;
  int j = j0 + threadIdx.x;
  if (j < TOPN_) {
    bx[threadIdx.x][0] = cand[j * 4 + 0];
    bx[threadIdx.x][1] = cand[j * 4 + 1];
    bx[threadIdx.x][2] = cand[j * 4 + 2];
    bx[threadIdx.x][3] = cand[j * 4 + 3];
  } else {
    bx[threadIdx.x][0] = 0.f; bx[threadIdx.x][1] = 0.f;
    bx[threadIdx.x][2] = 0.f; bx[threadIdx.x][3] = 0.f;
  }
  __syncthreads();
  if (i >= TOPN_) return;
  float x1 = cand[i * 4 + 0], y1 = cand[i * 4 + 1];
  float x2 = cand[i * 4 + 2], y2 = cand[i * 4 + 3];
  float ai = (x2 - x1) * (y2 - y1);
  unsigned long long m = 0ull;
#pragma unroll 4
  for (int t = 0; t < 64; ++t) {
    int jj = j0 + t;
    if (jj <= i || jj >= TOPN_) continue;
    float xx1 = fmaxf(x1, bx[t][0]);
    float yy1 = fmaxf(y1, bx[t][1]);
    float xx2 = fminf(x2, bx[t][2]);
    float yy2 = fminf(y2, bx[t][3]);
    float w = fmaxf(xx2 - xx1, 0.f), h = fmaxf(yy2 - yy1, 0.f);
    float inter = w * h;
    float aj = (bx[t][2] - bx[t][0]) * (bx[t][3] - bx[t][1]);
    float iou = inter / (ai + aj - inter + 1e-9f);
    if (iou > 0.7f) m |= (1ull << t);
  }
  mask[(size_t)i * NMSW + blockIdx.y] = m;
}

__global__ __launch_bounds__(128) void k_nms_reduce(const unsigned long long* __restrict__ mask,
                                                    const float* __restrict__ cand,
                                                    float* __restrict__ rois) {
  __shared__ unsigned long long rem[NMSW];
  __shared__ int keep[KEEPN];
  __shared__ int nk;
  int tid = threadIdx.x;
  for (int w = tid; w < NMSW; w += blockDim.x) rem[w] = 0ull;
  if (tid == 0) nk = 0;
  __syncthreads();
  for (int i = 0; i < TOPN_; ++i) {
    if (nk >= KEEPN) break;
    bool kp = ((rem[i >> 6] >> (i & 63)) & 1ull) == 0ull;
    __syncthreads();
    if (kp) {
      for (int w = tid; w < NMSW; w += blockDim.x) rem[w] |= mask[(size_t)i * NMSW + w];
      if (tid == 0) { keep[nk] = i; nk = nk + 1; }
    }
    __syncthreads();
  }
  __syncthreads();
  if (tid == 0) {
    int n = nk;
    for (int i = 0; i < TOPN_ && n < KEEPN; ++i)
      if ((rem[i >> 6] >> (i & 63)) & 1ull) keep[n++] = i;
    nk = n;
  }
  __syncthreads();
  for (int t = tid; t < KEEPN; t += blockDim.x) {
    int ki = (t < nk) ? keep[t] : 0;
    rois[t * 4 + 0] = cand[ki * 4 + 0];
    rois[t * 4 + 1] = cand[ki * 4 + 1];
    rois[t * 4 + 2] = cand[ki * 4 + 2];
    rois[t * 4 + 3] = cand[ki * 4 + 3];
  }
}

__global__ __launch_bounds__(256) void k_roi_align(const float* __restrict__ feat,
                                                   const float* __restrict__ rois,
                                                   _Float16* __restrict__ pool) {
  int r = blockIdx.x;
  float x1 = rois[r * 4 + 0] * 0.0625f, y1 = rois[r * 4 + 1] * 0.0625f;
  float x2 = rois[r * 4 + 2] * 0.0625f, y2 = rois[r * 4 + 3] * 0.0625f;
  float dx = (x2 - x1) * (1.f / 7.f), dy = (y2 - y1) * (1.f / 7.f);
  for (int e = threadIdx.x; e < POOLK; e += blockDim.x) {
    int c = e / 49, pe = e % 49, py = pe / 7, px = pe % 7;
    const float* f = feat + (size_t)c * NPOS;
    float acc = 0.f;
#pragma unroll
    for (int sy = 0; sy < 2; ++sy) {
#pragma unroll
      for (int sx = 0; sx < 2; ++sx) {
        float xs = x1 + ((px * 2 + sx) + 0.5f) * 0.5f * dx;
        float ys = y1 + ((py * 2 + sy) + 0.5f) * 0.5f * dy;
        float x0f = floorf(xs), y0f = floorf(ys);
        float lx = xs - x0f, ly = ys - y0f;
        int x0 = (int)fminf(fmaxf(x0f, 0.f), 49.f);
        int x1i = (int)fminf(fmaxf(x0f + 1.f, 0.f), 49.f);
        int y0 = (int)fminf(fmaxf(y0f, 0.f), 49.f);
        int y1i = (int)fminf(fmaxf(y0f + 1.f, 0.f), 49.f);
        float v00 = f[y0 * FW + x0],  v01 = f[y0 * FW + x1i];
        float v10 = f[y1i * FW + x0], v11 = f[y1i * FW + x1i];
        acc += v00 * (1.f - ly) * (1.f - lx) + v01 * (1.f - ly) * lx +
               v10 * ly * (1.f - lx) + v11 * ly * lx;
      }
    }
    pool[(size_t)r * POOLK + e] = (_Float16)(acc * 0.25f);
  }
}

__global__ void k_final(const float* __restrict__ cr, float* __restrict__ out) {
  int i = blockIdx.x * blockDim.x + threadIdx.x;
  if (i >= KEEPN * 105) return;
  int m = i / 105, n = i % 105;
  float v = cr[(size_t)m * 128 + n];
  if (n < 21) out[m * 21 + n] = v;
  else out[KEEPN * 21 + m * 84 + (n - 21)] = v;
}

// ---------------------------------------------------------------------------
static inline unsigned gblk(long long n, int b) { return (unsigned)((n + b - 1) / b); }

extern "C" void kernel_launch(void* const* d_in, const int* in_sizes, int n_in,
                              void* d_out, int out_size, void* d_ws, size_t ws_size,
                              hipStream_t stream) {
  const float* image   = (const float*)d_in[0];
  const float* w_feat  = (const float*)d_in[1];
  const float* b_feat  = (const float*)d_in[2];
  const float* w_rpn   = (const float*)d_in[3];
  const float* b_rpn   = (const float*)d_in[4];
  const float* w_score = (const float*)d_in[5];
  const float* b_score = (const float*)d_in[6];
  const float* w_delta = (const float*)d_in[7];
  const float* b_delta = (const float*)d_in[8];
  const float* w_fc7   = (const float*)d_in[9];
  const float* b_fc7   = (const float*)d_in[10];
  const float* w_cls   = (const float*)d_in[11];
  const float* b_cls   = (const float*)d_in[12];
  const float* w_reg   = (const float*)d_in[13];
  const float* b_reg   = (const float*)d_in[14];

  char* p = (char*)d_ws;
  auto carve = [&](size_t bytes) -> char* {
    char* r = p;
    p += (bytes + 255) & ~(size_t)255;
    return r;
  };

  _Float16* A1      = (_Float16*)carve((size_t)NPOS * K1 * 2);
  _Float16* wfeatT  = (_Float16*)carve((size_t)K1 * CF * 2);
  float*    featPC  = (float*)carve((size_t)NPOS * CF * 4);
  float*    featCHW = (float*)carve((size_t)CF * NPOS * 4);
  _Float16* A2      = (_Float16*)carve((size_t)NPOS * K2 * 2);
  _Float16* wrpnT   = (_Float16*)carve((size_t)K2 * CF * 2);
  _Float16* hbuf    = (_Float16*)carve((size_t)NPOS * CF * 2);
  _Float16* wsd     = (_Float16*)carve((size_t)CF * 64 * 2);
  float*    bias_sd = (float*)carve(64 * 4);
  float*    scdl    = (float*)carve((size_t)NPOS * 64 * 4);
  float*    fg      = (float*)carve((size_t)NANC * 4);
  float*    boxes   = (float*)carve((size_t)NANC * 4 * 4);
  float*    skey    = (float*)carve((size_t)SORTN * 4);
  int*      sidx    = (int*)carve((size_t)SORTN * 4);
  float*    cand    = (float*)carve((size_t)TOPN_ * 4 * 4);
  unsigned long long* mask = (unsigned long long*)carve((size_t)TOPN_ * NMSW * 8);
  float*    rois    = (float*)carve((size_t)KEEPN * 4 * 4);
  _Float16* pool    = (_Float16*)carve((size_t)KEEPN * POOLK * 2);
  float*    fc7p    = (float*)carve((size_t)FC7_KSPLIT * 320 * FC7N * 4);
  _Float16* fc7     = (_Float16*)carve((size_t)KEEPN * FC7N * 2);
  _Float16* wcr     = (_Float16*)carve((size_t)FC7N * 128 * 2);
  float*    bias_cr = (float*)carve(128 * 4);
  float*    outcr   = (float*)carve((size_t)KEEPN * 128 * 4);

  // --- weight prep + im2col (stem) ---
  k_im2col1<<<gblk((long long)NPOS * K1, 256), 256, 0, stream>>>(image, A1);
  k_wfeatT<<<gblk((long long)K1 * CF, 256), 256, 0, stream>>>(w_feat, wfeatT);
  k_wrpnT<<<gblk((long long)K2 * CF, 256), 256, 0, stream>>>(w_rpn, wrpnT);
  k_wsd<<<gblk((long long)CF * 64, 256), 256, 0, stream>>>(w_score, w_delta, b_score, b_delta, wsd, bias_sd);
  k_wcr<<<gblk((long long)FC7N * 128, 256), 256, 0, stream>>>(w_cls, w_reg, b_cls, b_reg, wcr, bias_cr);

  // --- G1: stem conv as GEMM [2500x768]x[768x512] -> featPC f32 ---
  gemm_wmma_f16<<<dim3(gblk(NPOS, 64), CF / 64), 256, 0, stream>>>(
      A1, K1, wfeatT, b_feat, 0, featPC, nullptr, CF, NPOS, CF, K1, CF);

  // --- feat repack: CHW for ROI-align, im2col for 3x3 RPN conv ---
  k_featT<<<gblk((long long)CF * NPOS, 256), 256, 0, stream>>>(featPC, featCHW);
  k_im2col2<<<gblk((long long)NPOS * K2, 256), 256, 0, stream>>>(featPC, A2);

  // --- G2: RPN 3x3 conv as GEMM [2500x4608]x[4608x512] + relu -> h f16 ---
  gemm_wmma_f16<<<dim3(gblk(NPOS, 64), CF / 64), 256, 0, stream>>>(
      A2, K2, wrpnT, b_rpn, 1, nullptr, hbuf, CF, NPOS, CF, K2, CF);

  // --- G3: score+delta GEMM [2500x512]x[512x64(54 real)] -> scdl f32 ---
  gemm_wmma_f16<<<dim3(gblk(NPOS, 64), 1), 256, 0, stream>>>(
      hbuf, CF, wsd, bias_sd, 0, scdl, nullptr, 64, NPOS, 64, CF, 54);

  // --- softmax + anchor decode ---
  k_rpn_post<<<gblk(NANC, 256), 256, 0, stream>>>(scdl, fg, boxes);

  // --- top-6000 via bitonic sort of padded 32768 keys (descending) ---
  k_sort_init<<<gblk(SORTN, 256), 256, 0, stream>>>(fg, skey, sidx);
  for (int k = 2; k <= SORTN; k <<= 1)
    for (int j = k >> 1; j > 0; j >>= 1)
      k_bitonic<<<SORTN / 256, 256, 0, stream>>>(skey, sidx, j, k);
  k_gather_cand<<<gblk(TOPN_, 256), 256, 0, stream>>>(boxes, sidx, cand);

  // --- NMS: 64x64 IoU bitmask tiles + one serial reduce block -> 300 rois ---
  k_nms_mask<<<dim3(NMSW, NMSW), 64, 0, stream>>>(cand, mask);
  k_nms_reduce<<<1, 128, 0, stream>>>(mask, cand, rois);

  // --- ROI align -> pool f16 [300][25088] ---
  k_roi_align<<<KEEPN, 256, 0, stream>>>(featCHW, rois, pool);

  // --- G4: fc7 GEMM, weights streamed once (split-K partials + reduce) ---
  gemm_fc7<<<dim3(FC7N / 64, FC7_KSPLIT), 256, 0, stream>>>(pool, w_fc7, fc7p);
  k_fc7_reduce<<<gblk((long long)KEEPN * FC7N, 256), 256, 0, stream>>>(fc7p, b_fc7, fc7);

  // --- G5: cls+reg GEMM [300x4096]x[4096x128(105 real)] -> outcr f32 ---
  gemm_wmma_f16<<<dim3(gblk(KEEPN, 64), 2), 256, 0, stream>>>(
      fc7, FC7N, wcr, bias_cr, 0, outcr, nullptr, 128, KEEPN, 128, FC7N, 105);

  // --- split into (score, bbox) layout of d_out ---
  k_final<<<gblk((long long)KEEPN * 105, 256), 256, 0, stream>>>(outcr, (float*)d_out);
}
